// MambaPython_33191507264051
// MI455X (gfx1250) — compile-verified, matrix operand order checked by reading, not run on hardware
//
#include <hip/hip_runtime.h>
#include <stdint.h>

#define D_MODEL 1024
#define HIDDEN  2048
#define D_STATE 16
#define BATCH   2
#define SEQ     2048
#define ROWS    (BATCH * SEQ)          // 4096

typedef __attribute__((ext_vector_type(16))) __bf16 v16bf;
typedef __attribute__((ext_vector_type(8)))  float  v8f;

union FragBF { uint4 q[2]; v16bf v; };

__device__ __forceinline__ void ldfrag(FragBF& f, const uint16_t* p) {
  f.q[0] = *(const uint4*)(p);
  f.q[1] = *(const uint4*)(p + 16);
}

__device__ __forceinline__ uint16_t f2bf(float f) {
  uint32_t u = __float_as_uint(f);
  u += 0x7FFFu + ((u >> 16) & 1u);       // round-to-nearest-even
  return (uint16_t)(u >> 16);
}

__device__ __forceinline__ float softplusf(float x) {
  return (x > 20.f) ? x : log1pf(__expf(x));
}

#define WMMA_BF16(A, B, C) \
  __builtin_amdgcn_wmma_f32_16x16x32_bf16(false, (A).v, false, (B).v, (short)0, (C), false, false)

// ---------------- f32 -> bf16 conversion ----------------
__global__ void k_f32_to_bf16(const float* __restrict__ src,
                              uint16_t* __restrict__ dst, int n) {
  int i = blockIdx.x * blockDim.x + threadIdx.x;
  if (i < n) dst[i] = f2bf(src[i]);
}

// ---------------- fused triple projection GEMM (bf16 WMMA, double-buffered) ----------------
// U = X@Wx^T + bx ; G = sigmoid(X@Wg^T + bg) ; DTpart = per-tile sum softplus(X@Wd^T + bd)
__global__ void k_proj_gemm(const uint16_t* __restrict__ Xbf,
                            const uint16_t* __restrict__ Wx,
                            const uint16_t* __restrict__ Wg,
                            const uint16_t* __restrict__ Wd,
                            const float* __restrict__ bx,
                            const float* __restrict__ bg,
                            const float* __restrict__ bd,
                            float* __restrict__ U,
                            float* __restrict__ G,
                            float* __restrict__ DTpart) {
  const int lane = threadIdx.x & 31;
  const int wave = threadIdx.x >> 5;
  const int tile = blockIdx.x * (blockDim.x >> 5) + wave;
  const int TN = HIDDEN / 16;            // 128 tiles along N
  const int tm = tile / TN, tn = tile % TN;
  const int m0 = tm * 16, n0 = tn * 16;
  const int l16 = lane & 15, hp = lane >> 4;   // half-wave select
  const int K = D_MODEL;

  v8f accU = {}, accG = {}, accD = {};
  const uint16_t* pa  = Xbf + (size_t)(m0 + l16) * K + hp * 8;
  const uint16_t* pwx = Wx  + (size_t)(n0 + l16) * K + hp * 8;
  const uint16_t* pwg = Wg  + (size_t)(n0 + l16) * K + hp * 8;
  const uint16_t* pwd = Wd  + (size_t)(n0 + l16) * K + hp * 8;

  FragBF a0, x0, g0, d0, a1, x1, g1, d1;
  ldfrag(a0, pa); ldfrag(x0, pwx); ldfrag(g0, pwg); ldfrag(d0, pwd);
  int k0 = 0;
  for (; k0 < K - 64; k0 += 64) {
    ldfrag(a1, pa + k0 + 32); ldfrag(x1, pwx + k0 + 32);
    ldfrag(g1, pwg + k0 + 32); ldfrag(d1, pwd + k0 + 32);
    __builtin_prefetch(pa + k0 + 256, 0, 1);
    accU = WMMA_BF16(a0, x0, accU);
    accG = WMMA_BF16(a0, g0, accG);
    accD = WMMA_BF16(a0, d0, accD);
    ldfrag(a0, pa + k0 + 64); ldfrag(x0, pwx + k0 + 64);
    ldfrag(g0, pwg + k0 + 64); ldfrag(d0, pwd + k0 + 64);
    accU = WMMA_BF16(a1, x1, accU);
    accG = WMMA_BF16(a1, g1, accG);
    accD = WMMA_BF16(a1, d1, accD);
  }
  ldfrag(a1, pa + k0 + 32); ldfrag(x1, pwx + k0 + 32);
  ldfrag(g1, pwg + k0 + 32); ldfrag(d1, pwd + k0 + 32);
  accU = WMMA_BF16(a0, x0, accU);
  accG = WMMA_BF16(a0, g0, accG);
  accD = WMMA_BF16(a0, d0, accD);
  accU = WMMA_BF16(a1, x1, accU);
  accG = WMMA_BF16(a1, g1, accG);
  accD = WMMA_BF16(a1, d1, accD);

  const int col = n0 + l16;
  float dsum = 0.f;
  const float bxc = bx[col], bgc = bg[col], bdc = bd[col];
#pragma unroll
  for (int r = 0; r < 8; ++r) {
    int row = m0 + r + hp * 8;
    size_t o = (size_t)row * HIDDEN + col;
    U[o] = accU[r] + bxc;
    float gl = accG[r] + bgc;
    G[o] = 1.0f / (1.0f + __expf(-gl));
    dsum += softplusf(accD[r] + bdc);
  }
  dsum += __shfl_xor(dsum, 16);          // combine the two 8-row halves (wave32)
  if (hp == 0) DTpart[(size_t)tm * HIDDEN + col] = dsum;
}

// ---------------- depthwise conv (k=3, same pad) + gating combine ----------------
// G <- U*g + (conv(U)+cb)*(1-g)
__global__ void k_conv_gate(const float* __restrict__ U,
                            float* __restrict__ G,
                            const float* __restrict__ cw,
                            const float* __restrict__ cb) {
  int idx = blockIdx.x * blockDim.x + threadIdx.x;
  if (idx >= ROWS * HIDDEN) return;
  int h = idx % HIDDEN;
  int t = (idx / HIDDEN) % SEQ;
  float um = (t > 0)       ? U[idx - HIDDEN] : 0.f;
  float uc = U[idx];
  float up = (t < SEQ - 1) ? U[idx + HIDDEN] : 0.f;
  float hv = cw[h * 3 + 0] * um + cw[h * 3 + 1] * uc + cw[h * 3 + 2] * up + cb[h];
  float g = G[idx];
  G[idx] = uc * g + hv * (1.f - g);
}

// ---------------- dt mean/clamp + SSM discretization ----------------
__global__ void k_dt_finalize(const float* __restrict__ DTpart,
                              const float* __restrict__ A_raw,
                              const float* __restrict__ B_ssm,
                              float* __restrict__ Ad,
                              float* __restrict__ Bd) {
  __shared__ float sm[256];
  int h = blockIdx.x;
  float s = 0.f;
  for (int i = threadIdx.x; i < ROWS / 16; i += blockDim.x)
    s += DTpart[(size_t)i * HIDDEN + h];
  sm[threadIdx.x] = s;
  __syncthreads();
  for (int o = 128; o > 0; o >>= 1) {
    if ((int)threadIdx.x < o) sm[threadIdx.x] += sm[threadIdx.x + o];
    __syncthreads();
  }
  float dt = sm[0] / (float)ROWS;
  dt = fminf(fmaxf(dt, 1e-3f), 2.0f);
  if (threadIdx.x < D_STATE) {
    int si = threadIdx.x;
    float A  = -softplusf(A_raw[h * D_STATE + si]);
    float ad = __expf(A * dt);
    float bd = (ad - 1.f) / (A + 1e-6f) * B_ssm[h * D_STATE + si];
    Ad[h * D_STATE + si] = ad;
    Bd[h * D_STATE + si] = bd;
  }
}

// ---------------- selective scan: one thread per (b,h) channel ----------------
__global__ void k_scan(const float* __restrict__ Uc,
                       const float* __restrict__ Ad,
                       const float* __restrict__ Bd,
                       const float* __restrict__ C,
                       uint16_t* __restrict__ Ybf) {
  int tid = blockIdx.x * blockDim.x + threadIdx.x;
  if (tid >= BATCH * HIDDEN) return;
  int h = tid % HIDDEN, b = tid / HIDDEN;
  float ad[D_STATE], bd[D_STATE], cc[D_STATE], st[D_STATE];
#pragma unroll
  for (int s = 0; s < D_STATE; ++s) {
    ad[s] = Ad[h * D_STATE + s];
    bd[s] = Bd[h * D_STATE + s];
    cc[s] = C[h * D_STATE + s];
    st[s] = 0.f;
  }
  size_t base = (size_t)b * SEQ * HIDDEN + h;
  for (int t = 0; t < SEQ; ++t) {
    float u = Uc[base + (size_t)t * HIDDEN];
    float p[4] = {0.f, 0.f, 0.f, 0.f};
#pragma unroll
    for (int s = 0; s < D_STATE; ++s) {
      st[s] = fmaf(ad[s], st[s], bd[s] * u);       // depth-1 across t, 16-way ILP
      p[s & 3] = fmaf(st[s], cc[s], p[s & 3]);     // 4 independent partials
    }
    float acc = (p[0] + p[1]) + (p[2] + p[3]);
    Ybf[base + (size_t)t * HIDDEN] = f2bf(acc);
  }
}

// ---------------- output GEMM (bf16 WMMA, 16x32 tile/wave, double-buffered) ----------------
__global__ void k_out_gemm(const uint16_t* __restrict__ Ybf,
                           const uint16_t* __restrict__ Wo,
                           const float* __restrict__ bo,
                           const float* __restrict__ xres,
                           float* __restrict__ out) {
  const int lane = threadIdx.x & 31;
  const int wave = threadIdx.x >> 5;
  const int tile = blockIdx.x * (blockDim.x >> 5) + wave;
  const int TG = D_MODEL / 32;           // 32 groups of two 16-wide N tiles
  const int tm = tile / TG, tg = tile % TG;
  const int m0 = tm * 16, n0 = tg * 32;
  const int l16 = lane & 15, hp = lane >> 4;
  const int K = HIDDEN;

  v8f acc0 = {}, acc1 = {};
  const uint16_t* pa  = Ybf + (size_t)(m0 + l16) * K + hp * 8;
  const uint16_t* pb0 = Wo  + (size_t)(n0 + l16) * K + hp * 8;
  const uint16_t* pb1 = Wo  + (size_t)(n0 + 16 + l16) * K + hp * 8;

  FragBF a0, c00, c10, a1, c01, c11;
  ldfrag(a0, pa); ldfrag(c00, pb0); ldfrag(c10, pb1);
  int k0 = 0;
  for (; k0 < K - 64; k0 += 64) {
    ldfrag(a1, pa + k0 + 32); ldfrag(c01, pb0 + k0 + 32); ldfrag(c11, pb1 + k0 + 32);
    __builtin_prefetch(pa + k0 + 256, 0, 1);
    acc0 = WMMA_BF16(a0, c00, acc0);
    acc1 = WMMA_BF16(a0, c10, acc1);
    ldfrag(a0, pa + k0 + 64); ldfrag(c00, pb0 + k0 + 64); ldfrag(c10, pb1 + k0 + 64);
    acc0 = WMMA_BF16(a1, c01, acc0);
    acc1 = WMMA_BF16(a1, c11, acc1);
  }
  ldfrag(a1, pa + k0 + 32); ldfrag(c01, pb0 + k0 + 32); ldfrag(c11, pb1 + k0 + 32);
  acc0 = WMMA_BF16(a0, c00, acc0);
  acc1 = WMMA_BF16(a0, c10, acc1);
  acc0 = WMMA_BF16(a1, c01, acc0);
  acc1 = WMMA_BF16(a1, c11, acc1);

  const int col0 = n0 + l16, col1 = n0 + 16 + l16;
  const float bo0 = bo[col0], bo1 = bo[col1];
#pragma unroll
  for (int r = 0; r < 8; ++r) {
    int row = m0 + r + hp * 8;
    size_t o0 = (size_t)row * D_MODEL + col0;
    size_t o1 = (size_t)row * D_MODEL + col1;
    out[o0] = acc0[r] + bo0 + xres[o0];
    out[o1] = acc1[r] + bo1 + xres[o1];
  }
}

// ---------------- in-place layernorm over D_MODEL ----------------
__global__ void k_layernorm(float* __restrict__ out,
                            const float* __restrict__ g,
                            const float* __restrict__ b) {
  __shared__ float s1[256], s2[256];
  int row = blockIdx.x;
  float* p = out + (size_t)row * D_MODEL;
  float sum = 0.f, sq = 0.f;
  for (int i = threadIdx.x; i < D_MODEL; i += blockDim.x) {
    float v = p[i]; sum += v; sq += v * v;
  }
  s1[threadIdx.x] = sum; s2[threadIdx.x] = sq;
  __syncthreads();
  for (int o = 128; o > 0; o >>= 1) {
    if ((int)threadIdx.x < o) {
      s1[threadIdx.x] += s1[threadIdx.x + o];
      s2[threadIdx.x] += s2[threadIdx.x + o];
    }
    __syncthreads();
  }
  float m = s1[0] / (float)D_MODEL;
  float var = s2[0] / (float)D_MODEL - m * m;
  float r = rsqrtf(var + 1e-5f);
  for (int i = threadIdx.x; i < D_MODEL; i += blockDim.x) {
    float v = p[i];
    p[i] = (v - m) * r * g[i] + b[i];
  }
}

extern "C" void kernel_launch(void* const* d_in, const int* in_sizes, int n_in,
                              void* d_out, int out_size, void* d_ws, size_t ws_size,
                              hipStream_t stream) {
  const float* x      = (const float*)d_in[0];
  const float* W_x    = (const float*)d_in[1];
  const float* b_x    = (const float*)d_in[2];
  const float* W_g    = (const float*)d_in[3];
  const float* b_g    = (const float*)d_in[4];
  const float* conv_w = (const float*)d_in[5];
  const float* conv_b = (const float*)d_in[6];
  const float* A_raw  = (const float*)d_in[7];
  const float* B_ssm  = (const float*)d_in[8];
  const float* C_ssm  = (const float*)d_in[9];
  const float* W_dt   = (const float*)d_in[10];
  const float* b_dt   = (const float*)d_in[11];
  const float* W_out  = (const float*)d_in[12];
  const float* b_out  = (const float*)d_in[13];
  const float* ln_g   = (const float*)d_in[14];
  const float* ln_b   = (const float*)d_in[15];
  float* out = (float*)d_out;

  uint8_t* ws = (uint8_t*)d_ws;
  size_t off = 0;
  auto carve = [&](size_t bytes) -> void* {
    void* p = ws + off;
    off += (bytes + 255) & ~(size_t)255;
    return p;
  };
  uint16_t* Xbf    = (uint16_t*)carve((size_t)ROWS * D_MODEL * 2);
  uint16_t* Wxbf   = (uint16_t*)carve((size_t)HIDDEN * D_MODEL * 2);
  uint16_t* Wgbf   = (uint16_t*)carve((size_t)HIDDEN * D_MODEL * 2);
  uint16_t* Wdbf   = (uint16_t*)carve((size_t)HIDDEN * D_MODEL * 2);
  uint16_t* Wobf   = (uint16_t*)carve((size_t)D_MODEL * HIDDEN * 2);
  float*    U      = (float*)carve((size_t)ROWS * HIDDEN * 4);
  float*    G      = (float*)carve((size_t)ROWS * HIDDEN * 4);
  uint16_t* Ybf    = (uint16_t*)carve((size_t)ROWS * HIDDEN * 2);
  float*    DTpart = (float*)carve((size_t)(ROWS / 16) * HIDDEN * 4);
  float*    Ad     = (float*)carve((size_t)HIDDEN * D_STATE * 4);
  float*    Bd     = (float*)carve((size_t)HIDDEN * D_STATE * 4);
  (void)ws_size; (void)in_sizes; (void)n_in; (void)out_size;

  // 1) precision conversion to bf16
  int nX = ROWS * D_MODEL, nW = HIDDEN * D_MODEL;
  k_f32_to_bf16<<<(nX + 255) / 256, 256, 0, stream>>>(x, Xbf, nX);
  k_f32_to_bf16<<<(nW + 255) / 256, 256, 0, stream>>>(W_x, Wxbf, nW);
  k_f32_to_bf16<<<(nW + 255) / 256, 256, 0, stream>>>(W_g, Wgbf, nW);
  k_f32_to_bf16<<<(nW + 255) / 256, 256, 0, stream>>>(W_dt, Wdbf, nW);
  k_f32_to_bf16<<<(nW + 255) / 256, 256, 0, stream>>>(W_out, Wobf, nW);

  // 2) fused triple projection GEMM: (ROWS/16)*(HIDDEN/16)=32768 tiles, 8 waves/block
  k_proj_gemm<<<(ROWS / 16) * (HIDDEN / 16) / 8, 256, 0, stream>>>(
      Xbf, Wxbf, Wgbf, Wdbf, b_x, b_g, b_dt, U, G, DTpart);

  // 3) depthwise conv + gate combine (writes combined u into G)
  k_conv_gate<<<(ROWS * HIDDEN + 255) / 256, 256, 0, stream>>>(U, G, conv_w, conv_b);

  // 4) dt reduction + discretization
  k_dt_finalize<<<HIDDEN, 256, 0, stream>>>(DTpart, A_raw, B_ssm, Ad, Bd);

  // 5) selective scan -> y (bf16)
  k_scan<<<(BATCH * HIDDEN + 255) / 256, 256, 0, stream>>>(G, Ad, Bd, C_ssm, Ybf);

  // 6) output GEMM (16x32 tiles) + bias + residual -> d_out
  k_out_gemm<<<(ROWS / 16) * (D_MODEL / 32) / 8, 256, 0, stream>>>(
      Ybf, Wobf, b_out, x, out);

  // 7) in-place layernorm
  k_layernorm<<<ROWS, 256, 0, stream>>>(out, ln_g, ln_b);
}
